// RGCNConvCuGraph_45724221833502
// MI455X (gfx1250) — compile-verified, matrix operand order checked by reading, not run on hardware
//
#include <hip/hip_runtime.h>

// ---------------------------------------------------------------------------
// RGCN (basis decomposition) for MI455X / gfx1250 — fused LDS version.
//
//   Wrel[r] = sum_b comp[r,b] * weight[b]   (folded once; root = weight[8])
//   CSR-sort edges by destination (integer atomics only), then per 16-node
//   tile: aggregate x[src] into a [16, 16*128] f32 accumulator in LDS
//   (128 KB of the WGP's 320 KB), and run the GEMM
//       out = inv_deg * (h @ Wrel_stack) + x @ W_root + bias
//   with v_wmma_f32_16x16x32_bf16 fed directly from LDS. No FP atomics,
//   no 410 MB h round-trip.
// ---------------------------------------------------------------------------

typedef __attribute__((ext_vector_type(16))) __bf16 v16bf;
typedef __attribute__((ext_vector_type(8)))  float  v8f;

constexpr int N_NODES   = 100000;
constexpr int N_EDGES   = 1600000;
constexpr int IN_CH     = 128;
constexpr int OUT_CH    = 128;
constexpr int NUM_REL   = 16;
constexpr int NUM_BASES = 8;

constexpr int KH     = NUM_REL * IN_CH;   // 2048 (relation-stacked K)
constexpr int KT_H   = KH / 32;           // 64 k-tiles for h
constexpr int KT_X   = IN_CH / 32;        // 4 k-tiles for root
constexpr int KT_TOT = KT_H + KT_X;       // 68
constexpr int CT     = OUT_CH / 16;       // 8 column tiles

constexpr int HPAD = KH + 4;              // f32 LDS row stride (bank spread)
constexpr size_t LDS_BYTES = (size_t)16 * HPAD * sizeof(float);  // 131,328 B

// workspace layout (bytes), all 16B-aligned
constexpr size_t XBF_BYTES  = (size_t)N_NODES * IN_CH * 2;        // 25.6 MB
constexpr size_t BP_BYTES   = (size_t)CT * KT_TOT * 32 * 16 * 2;  // 557 KB
constexpr size_t CNT_BYTES  = (size_t)N_NODES * 4;
constexpr size_t ROW_BYTES  = (size_t)N_NODES * 4;
constexpr size_t FILL_BYTES = (size_t)N_NODES * 4;

__device__ __forceinline__ unsigned int f2bf(float f) {        // RNE-ish
  unsigned int u = __builtin_bit_cast(unsigned int, f);
  u += 0x7FFFu + ((u >> 16) & 1u);
  return (u >> 16) & 0xFFFFu;
}
__device__ __forceinline__ unsigned int pkbf(float a, float b) { // round-half-up pack
  unsigned int ua = __builtin_bit_cast(unsigned int, a) + 0x8000u;
  unsigned int ub = __builtin_bit_cast(unsigned int, b) + 0x8000u;
  return (ua >> 16) | (ub & 0xFFFF0000u);
}
__device__ __forceinline__ uint4 pack8(const float* p) {  // 8 f32 -> 8 bf16
  float4 f0 = *(const float4*)p;
  float4 f1 = *(const float4*)(p + 4);
  uint4 r;
  r.x = pkbf(f0.x, f0.y); r.y = pkbf(f0.z, f0.w);
  r.z = pkbf(f1.x, f1.y); r.w = pkbf(f1.z, f1.w);
  return r;
}

// --- CSR build: count -> exclusive scan -> slot scatter ---------------------
__global__ void count_edges(const int* __restrict__ edge_index, int* __restrict__ cnt) {
  int t = blockIdx.x * blockDim.x + threadIdx.x;
  if (t < N_EDGES) atomicAdd(&cnt[edge_index[N_EDGES + t]], 1);
}

__global__ void __launch_bounds__(1024)
scan_excl(const int* __restrict__ cnt, int* __restrict__ rowstart) {
  __shared__ int buf[1024];
  const int tid = threadIdx.x;
  int carry = 0;
  for (int base = 0; base < N_NODES; base += 1024) {
    int idx = base + tid;
    int v = (idx < N_NODES) ? cnt[idx] : 0;
    buf[tid] = v;
    __syncthreads();
    int val = v;
    for (int off = 1; off < 1024; off <<= 1) {
      int t = (tid >= off) ? buf[tid - off] : 0;
      __syncthreads();
      val += t;
      buf[tid] = val;
      __syncthreads();
    }
    if (idx < N_NODES) rowstart[idx] = carry + val - v;   // exclusive
    carry += buf[1023];
    __syncthreads();
  }
}

__global__ void scatter_sort(const int* __restrict__ edge_index,
                             const int* __restrict__ edge_type,
                             int* __restrict__ fill, unsigned int* __restrict__ epack) {
  int t = blockIdx.x * blockDim.x + threadIdx.x;
  if (t >= N_EDGES) return;
  int src = edge_index[t];
  int dst = edge_index[N_EDGES + t];
  int et  = edge_type[t];
  int pos = atomicAdd(&fill[dst], 1);
  epack[pos] = ((unsigned int)et << 17) | (unsigned int)src;   // src < 2^17
}

// --- prep: fold comp into weights, pack into WMMA-B lane layout -------------
// B tile (32x16 bf16) for (ct,kt): lane n<16 holds W[k0+{0..7,16..23}][col],
// lane n+16 holds W[k0+{8..15,24..31}][col]; 16 bf16 contiguous per lane.
__global__ void prep_wpack(const float* __restrict__ weight,
                           const float* __restrict__ comp,
                           unsigned short* __restrict__ bpack) {
  int t = blockIdx.x * blockDim.x + threadIdx.x;
  if (t >= CT * KT_TOT * 32) return;
  int lane = t & 31;
  int kt   = (t >> 5) % KT_TOT;
  int ct   = t / (KT_TOT * 32);
  int half = lane >> 4;
  int col  = ct * 16 + (lane & 15);

  unsigned short vals[16];
#pragma unroll
  for (int i = 0; i < 16; ++i) {
    int kl = (i >> 3) * 16 + half * 8 + (i & 7);
    int kg = kt * 32 + kl;                        // 0..2175
    float v;
    if (kg < KH) {
      int r = kg >> 7, kk = kg & 127;
      float acc = 0.f;
#pragma unroll
      for (int b = 0; b < NUM_BASES; ++b)
        acc += comp[r * NUM_BASES + b] * weight[((size_t)b * 128 + kk) * 128 + col];
      v = acc;
    } else {
      int kk = kg - KH;                            // root slot (weight[8])
      v = weight[((size_t)NUM_BASES * 128 + kk) * 128 + col];
    }
    vals[i] = (unsigned short)f2bf(v);
  }
  unsigned short* dst = bpack + ((size_t)(ct * KT_TOT + kt) * 32 + lane) * 16;
#pragma unroll
  for (int i = 0; i < 16; ++i) dst[i] = vals[i];
}

__global__ void prep_xbf(const float* __restrict__ x, unsigned short* __restrict__ xbf) {
  int t = blockIdx.x * blockDim.x + threadIdx.x;   // one float4 per thread
  float4 v = ((const float4*)x)[t];
  uint2 o;
  o.x = f2bf(v.x) | (f2bf(v.y) << 16);
  o.y = f2bf(v.z) | (f2bf(v.w) << 16);
  ((uint2*)xbf)[t] = o;
}

// --- fused aggregate (LDS, f32) + WMMA GEMM ---------------------------------
union V16 { uint4 q[2]; v16bf v; };

__global__ void __launch_bounds__(256)
rgcn_fused(const float* __restrict__ x, const unsigned int* __restrict__ epack,
           const int* __restrict__ rowstart, const int* __restrict__ cnt,
           const unsigned short* __restrict__ xbf, const unsigned short* __restrict__ bpack,
           const float* __restrict__ bias, float* __restrict__ out) {
  extern __shared__ float hacc[];                  // [16][HPAD] f32
  const int tile = blockIdx.x;                     // 16-node row tile
  const int wave = threadIdx.x >> 5;               // 0..7
  const int lane = threadIdx.x & 31;

  // zero the accumulator tile
  for (int i = threadIdx.x; i < 16 * HPAD; i += 256) hacc[i] = 0.f;
  __syncthreads();

  // phase 1: each wave owns nodes {wave, wave+8}; sequential edge walk,
  // plain LDS read-modify-write (no atomics needed — single owner).
  for (int m = wave; m < 16; m += 8) {
    const int node = tile * 16 + m;
    const int beg  = rowstart[node];
    const int num  = cnt[node];
    float* hrow = hacc + m * HPAD;
    for (int j = 0; j < num; ++j) {
      unsigned int rec = epack[beg + j];
      int src = rec & 0x1FFFFu;
      int r   = rec >> 17;
      float4 xv = *((const float4*)(x + (size_t)src * IN_CH) + lane);  // 4 ch/lane
      float* p = hrow + r * IN_CH + lane * 4;
      float4 cur = *(float4*)p;
      cur.x += xv.x; cur.y += xv.y; cur.z += xv.z; cur.w += xv.w;
      *(float4*)p = cur;
    }
  }
  __syncthreads();

  // phase 2: WMMA. wave = column tile. A 16x32 bf16 lane layout:
  //   lane<16 -> K {k0..k0+7, k0+16..k0+23}, lane>=16 -> +8 on both groups.
  const int half = lane >> 4;
  const int rn   = lane & 15;
  v8f acc_h = {};    // relation part (gets 1/deg)
  v8f acc_r = {};    // root part
  const unsigned short* bbase = bpack + (size_t)wave * KT_TOT * 512;

#pragma unroll 4
  for (int kt = 0; kt < KT_H; ++kt) {
    const int k0 = kt * 32 + half * 8;
    const float* hr = hacc + rn * HPAD + k0;
    V16 a, b;
    a.q[0] = pack8(hr);          // f32 LDS -> bf16, co-executes with WMMA
    a.q[1] = pack8(hr + 16);
    const unsigned short* bt = bbase + (size_t)kt * 512 + lane * 16;
    b.q[0] = *(const uint4*)(bt);
    b.q[1] = *(const uint4*)(bt + 8);
    acc_h = __builtin_amdgcn_wmma_f32_16x16x32_bf16(false, a.v, false, b.v,
                                                    (short)0, acc_h, false, false);
  }

  const unsigned short* xrow = xbf + (size_t)(tile * 16 + rn) * IN_CH;
#pragma unroll
  for (int kt = 0; kt < KT_X; ++kt) {
    const int k0 = kt * 32 + half * 8;
    V16 a, b;
    a.q[0] = *(const uint4*)(xrow + k0);
    a.q[1] = *(const uint4*)(xrow + k0 + 16);
    const unsigned short* bt = bbase + (size_t)(KT_H + kt) * 512 + lane * 16;
    b.q[0] = *(const uint4*)(bt);
    b.q[1] = *(const uint4*)(bt + 8);
    acc_r = __builtin_amdgcn_wmma_f32_16x16x32_bf16(false, a.v, false, b.v,
                                                    (short)0, acc_r, false, false);
  }

  // epilogue. C layout: lane<16 VGPR v -> (M=v, N=lane); lane>=16 -> (M=v+8).
  const int col = wave * 16 + rn;
  const float bc = bias[col];
#pragma unroll
  for (int v = 0; v < 8; ++v) {
    int node = tile * 16 + v + half * 8;
    float inv = 1.0f / fmaxf((float)cnt[node], 1.0f);
    out[(size_t)node * OUT_CH + col] = inv * acc_h[v] + acc_r[v] + bc;
  }
}

// ---------------------------------------------------------------------------
extern "C" void kernel_launch(void* const* d_in, const int* in_sizes, int n_in,
                              void* d_out, int out_size, void* d_ws, size_t ws_size,
                              hipStream_t stream) {
  const float* x          = (const float*)d_in[0];
  const float* weight     = (const float*)d_in[1];
  const float* comp       = (const float*)d_in[2];
  const float* bias       = (const float*)d_in[3];
  const int*   edge_index = (const int*)d_in[4];
  const int*   edge_type  = (const int*)d_in[5];
  float*       out        = (float*)d_out;

  char* ws = (char*)d_ws;
  unsigned short* xbf      = (unsigned short*)ws;
  unsigned short* bpack    = (unsigned short*)(ws + XBF_BYTES);
  int*            cnt      = (int*)(ws + XBF_BYTES + BP_BYTES);
  int*            rowstart = (int*)(ws + XBF_BYTES + BP_BYTES + CNT_BYTES);
  int*            fill     = (int*)(ws + XBF_BYTES + BP_BYTES + CNT_BYTES + ROW_BYTES);
  unsigned int*   epack    = (unsigned int*)(ws + XBF_BYTES + BP_BYTES + CNT_BYTES +
                                             ROW_BYTES + FILL_BYTES);

  // allow >64KB dynamic LDS for the fused kernel (host-side, idempotent)
  (void)hipFuncSetAttribute((const void*)rgcn_fused,
                            hipFuncAttributeMaxDynamicSharedMemorySize,
                            (int)LDS_BYTES);

  hipMemsetAsync(cnt, 0, CNT_BYTES, stream);
  count_edges<<<N_EDGES / 256, 256, 0, stream>>>(edge_index, cnt);
  scan_excl<<<1, 1024, 0, stream>>>(cnt, rowstart);
  hipMemcpyAsync(fill, rowstart, ROW_BYTES, hipMemcpyDeviceToDevice, stream);
  scatter_sort<<<N_EDGES / 256, 256, 0, stream>>>(edge_index, edge_type, fill, epack);

  prep_wpack<<<(CT * KT_TOT * 32 + 255) / 256, 256, 0, stream>>>(weight, comp, bpack);
  prep_xbf<<<(N_NODES * IN_CH / 4) / 256, 256, 0, stream>>>(x, xbf);

  rgcn_fused<<<N_NODES / 16, 256, LDS_BYTES, stream>>>(x, epack, rowstart, cnt,
                                                       xbf, bpack, bias, out);
}
